// LSTM_51728586113045
// MI455X (gfx1250) — matrix operand order repaired and checked by reading, not compiled, not run on hardware
//
#include <hip/hip_runtime.h>

typedef __attribute__((ext_vector_type(16))) __bf16    v16bf;
typedef __attribute__((ext_vector_type(2)))  __bf16    v2bf;
typedef __attribute__((ext_vector_type(8)))  float     v8f;
typedef __attribute__((ext_vector_type(8)))  unsigned  v8u;

__device__ inline v8f vzero8f() {
  v8f z;
#pragma unroll
  for (int i = 0; i < 8; ++i) z[i] = 0.f;
  return z;
}

__device__ inline v8u vzero8u() {
  v8u z;
#pragma unroll
  for (int i = 0; i < 8; ++i) z[i] = 0u;
  return z;
}

// f32 -> packed bf16 pair (a in [15:0], b in [31:16]).
// Prefer the dedicated builtin; else native fptrunc (backend picks the best
// instruction, e.g. v_cvt_pk_bf16_f32 if present).
__device__ inline unsigned pack_bf16(float a, float b) {
#if __has_builtin(__builtin_amdgcn_cvt_pk_bf16_f32)
  return __builtin_bit_cast(unsigned, __builtin_amdgcn_cvt_pk_bf16_f32(a, b));
#else
  v2bf p;
  p[0] = (__bf16)a;
  p[1] = (__bf16)b;
  return __builtin_bit_cast(unsigned, p);
#endif
}

__device__ inline v8f wmma_bf16(const v8u a, const v8u b, const v8f c) {
  return __builtin_amdgcn_wmma_f32_16x16x32_bf16(
      /*neg_a=*/false, __builtin_bit_cast(v16bf, a),
      /*neg_b=*/false, __builtin_bit_cast(v16bf, b),
      /*c_mod=*/(short)0, c, /*reuse_a=*/false, /*reuse_b=*/false);
}

// Hardware tanh (confirmed: v_tanh_f32 on gfx1250), fallbacks kept for safety.
__device__ inline float tanh_f(float x) {
#if __has_builtin(__builtin_amdgcn_tanhf)
  return __builtin_amdgcn_tanhf(x);
#elif __has_builtin(__builtin_amdgcn_tanh_f32)
  return __builtin_amdgcn_tanh_f32(x);
#else
  return 1.f - 2.f * __builtin_amdgcn_rcpf(1.f + __expf(2.f * x));
#endif
}

__device__ inline float sigm(float x) {
#if __has_builtin(__builtin_amdgcn_tanhf) || __has_builtin(__builtin_amdgcn_tanh_f32)
  return fmaf(0.5f, tanh_f(0.5f * x), 0.5f);  // sigmoid via hw tanh
#else
  return __builtin_amdgcn_rcpf(1.f + __expf(-x));
#endif
}

// A-operand (16x32 bf16) tile loader: row-major f32 weights, K padded with 0.
// A layout: lane m=lane&15 holds row m; VGPR v holds K pair
//   k0 = (v<4 ? 2v : 16+2(v-4)) + (lane>>4)*8
template <int KMAX>
__device__ inline v8u load_wA(const float* __restrict__ W, int row, int ldk, int sel) {
  const float* p = W + (size_t)row * ldk;
  v8u r;
#pragma unroll
  for (int v = 0; v < 8; ++v) {
    const int k0 = (v < 4 ? 2 * v : 16 + 2 * (v - 4)) + sel * 8;
    float a, b;
    if constexpr (KMAX >= 32) {
      const float2 q = *(const float2*)(p + k0);
      a = q.x; b = q.y;
    } else {
      a = (k0     < KMAX) ? p[k0]     : 0.f;
      b = (k0 + 1 < KMAX) ? p[k0 + 1] : 0.f;
    }
    r[v] = pack_bf16(a, b);
  }
  return r;
}

// Elementwise LSTM cell update on 16-hidden x 16-batch tiles (D layout).
__device__ inline v8f lstm_cell(v8f& c, const v8f ig, const v8f fg,
                                const v8f gg, const v8f og) {
  v8f h;
#pragma unroll
  for (int e = 0; e < 8; ++e) {
    const float i = sigm(ig[e]), f = sigm(fg[e]);
    const float g = tanh_f(gg[e]), o = sigm(og[e]);
    const float cn = f * c[e] + i * g;
    c[e] = cn;
    h[e] = o * tanh_f(cn);
  }
  return h;
}

// Rebuild B operand (32x16 bf16, hidden-K x batch-N) from h tiles in D layout.
// No cross-half move: D-layout pairs already fit B; the ds_swizzle (0x401F, xor 0x10) results are unused.
__device__ inline v8u make_Bh(const v8f hlo, const v8f hhi, bool low) {
  v8u B;
#pragma unroll
  for (int v = 0; v < 4; ++v) {
    const unsigned plo = pack_bf16(hlo[2 * v], hlo[2 * v + 1]);
    const unsigned phi = pack_bf16(hhi[2 * v], hhi[2 * v + 1]);
    const unsigned slo = (unsigned)__builtin_amdgcn_ds_swizzle((int)plo, 0x401f);
    const unsigned shi = (unsigned)__builtin_amdgcn_ds_swizzle((int)phi, 0x401f);
    B[v]     = plo;
    B[v + 4] = phi;
  }
  return B;
}

__global__ __launch_bounds__(32, 1) void lstm2_fc_softmax_fused(
    const float* __restrict__ x,
    const float* __restrict__ wih0, const float* __restrict__ whh0,
    const float* __restrict__ bih0, const float* __restrict__ bhh0,
    const float* __restrict__ wih1, const float* __restrict__ whh1,
    const float* __restrict__ bih1, const float* __restrict__ bhh1,
    const float* __restrict__ wfc,  const float* __restrict__ bfc,
    float* __restrict__ out) {
  constexpr int T = 128, I = 9, H = 32;
  const int lane = (int)threadIdx.x;
  const int m    = lane & 15;
  const int sel  = lane >> 4;
  const bool low = (sel == 0);
  const int b0   = blockIdx.x * 16;  // 16 batch rows per wave

  // ---- register-resident weights (A operands) and bias-preloaded C tiles ----
  v8u Aih0[8], Ahh0[8], Aih1[8], Ahh1[8];
  v8f C0[8], C1[8];
#pragma unroll
  for (int g = 0; g < 8; ++g) {
    const int row = g * 16 + m;  // gate index 0..127 (i,f,g,o stacked)
    Aih0[g] = load_wA<9 >(wih0, row, I, sel);
    Ahh0[g] = load_wA<32>(whh0, row, H, sel);
    Aih1[g] = load_wA<32>(wih1, row, H, sel);
    Ahh1[g] = load_wA<32>(whh1, row, H, sel);
#pragma unroll
    for (int j = 0; j < 8; ++j) {
      const int idx = g * 16 + j + sel * 8;  // D layout: VGPR j -> gate row j / j+8
      C0[g][j] = bih0[idx] + bhh0[idx];
      C1[g][j] = bih1[idx] + bhh1[idx];
    }
  }

  // ---- persistent state ----
  v8f c0lo = vzero8f(), c0hi = vzero8f();  // layer0 cell state (hid 0-15 / 16-31)
  v8f c1lo = vzero8f(), c1hi = vzero8f();  // layer1 cell state
  v8u Bh1 = vzero8u();                     // h1_{t-1} as B operand (layer0 recur.)
  v8u Bh2 = vzero8u();                     // h2_{t-1} as B operand (layer1 recur.)
  v8f acc = vzero8f();                     // FC logits D tile [o x batch]

  const float* xr = x + (size_t)(b0 + m) * (T * I);

#pragma unroll 1
  for (int t = 0; t < T; ++t) {
    // ---- B operand from x[:, t, :]  (K = input feature, zero-padded 9->32) ----
    v8u Bx;
#pragma unroll
    for (int v = 0; v < 8; ++v) {
      const int k0 = (v < 4 ? 2 * v : 16 + 2 * (v - 4)) + sel * 8;
      const float va = (k0     < I) ? xr[t * I + k0]     : 0.f;
      const float vb = (k0 + 1 < I) ? xr[t * I + k0 + 1] : 0.f;
      Bx[v] = pack_bf16(va, vb);
    }
    __builtin_prefetch(xr + (t + 1) * I, 0, 1);  // global_prefetch_b8

    // ---- layer 0: gates = bias + w_ih0*x_t + w_hh0*h1_{t-1} ----
    v8f D[8];
#pragma unroll
    for (int g = 0; g < 8; ++g) {
      D[g] = wmma_bf16(Aih0[g], Bx, C0[g]);
      D[g] = wmma_bf16(Ahh0[g], Bh1, D[g]);
    }
    const v8f h1lo = lstm_cell(c0lo, D[0], D[2], D[4], D[6]);
    const v8f h1hi = lstm_cell(c0hi, D[1], D[3], D[5], D[7]);
    Bh1 = make_Bh(h1lo, h1hi, low);  // feeds layer1 now and layer0 next step

    // ---- layer 1: gates = bias + w_ih1*h1_t + w_hh1*h2_{t-1} ----
#pragma unroll
    for (int g = 0; g < 8; ++g) {
      D[g] = wmma_bf16(Aih1[g], Bh1, C1[g]);
      D[g] = wmma_bf16(Ahh1[g], Bh2, D[g]);
    }
    const v8f h2lo = lstm_cell(c1lo, D[0], D[2], D[4], D[6]);
    const v8f h2hi = lstm_cell(c1hi, D[1], D[3], D[5], D[7]);
    Bh2 = make_Bh(h2lo, h2hi, low);  // feeds FC now and layer1 next step

    // ---- FC accumulate: logits += w_fc[:, t*32:(t+1)*32] * h2_t^T ----
    v8u Afc;
    const int orow = (m < 6) ? m : 5;  // clamp padded rows into valid memory
    const float* pf = wfc + (size_t)orow * (T * H) + t * H;
#pragma unroll
    for (int v = 0; v < 8; ++v) {
      const int k0 = (v < 4 ? 2 * v : 16 + 2 * (v - 4)) + sel * 8;
      const float2 q = *(const float2*)(pf + k0);
      Afc[v] = pack_bf16(q.x, q.y);
    }
    acc = wmma_bf16(Afc, Bh2, acc);
  }

  // ---- epilogue: bias + ReLU + softmax over the 6 valid logits ----
  if (low) {  // lanes 0..15: lane = batch n, VGPR j = output j (j<6 valid)
    const int b = b0 + m;
    float l[6], mx = 0.f;  // ReLU floor => max >= 0
#pragma unroll
    for (int j = 0; j < 6; ++j) {
      float v = acc[j] + bfc[j];
      v = v > 0.f ? v : 0.f;
      l[j] = v;
      mx = v > mx ? v : mx;
    }
    float e[6], s = 0.f;
#pragma unroll
    for (int j = 0; j < 6; ++j) { e[j] = __expf(l[j] - mx); s += e[j]; }
#pragma unroll
    for (int j = 0; j < 6; ++j) out[(size_t)b * 6 + j] = e[j] / s;
  }
}

extern "C" void kernel_launch(void* const* d_in, const int* in_sizes, int n_in,
                              void* d_out, int out_size, void* d_ws, size_t ws_size,
                              hipStream_t stream) {
  const float* x    = (const float*)d_in[0];
  const float* wih0 = (const float*)d_in[1];
  const float* whh0 = (const float*)d_in[2];
  const float* bih0 = (const float*)d_in[3];
  const float* bhh0 = (const float*)d_in[4];
  const float* wih1 = (const float*)d_in[5];
  const float* whh1 = (const float*)d_in[6];
  const float* bih1 = (const float*)d_in[7];
  const float* bhh1 = (const float*)d_in[8];
  const float* wfc  = (const float*)d_in[9];
  const float* bfc  = (const float*)d_in[10];

  const int B = in_sizes[0] / (128 * 9);  // 8192
  const int nblk = B / 16;                // one 16-row batch tile per wave

  lstm2_fc_softmax_fused<<<nblk, 32, 0, stream>>>(
      x, wih0, whh0, bih0, bhh0, wih1, whh1, bih1, bhh1, wfc, bfc,
      (float*)d_out);
}
